// GraphEnhancedLayoutLMv3_71725953843705
// MI455X (gfx1250) — matrix operand order repaired and checked by reading, not compiled, not run on hardware
//
#include <hip/hip_runtime.h>
#include <hip/hip_bf16.h>

// ---------------------------------------------------------------------------
// Types for CDNA5 WMMA (wave32): v_wmma_f32_16x16x32_bf16
// ---------------------------------------------------------------------------
typedef __bf16 bf16_t;
typedef __attribute__((ext_vector_type(16))) __bf16 v16bf;
typedef __attribute__((ext_vector_type(8)))  float  v8f;
typedef __attribute__((ext_vector_type(4)))  int    v4i;

#define H 768
#define LAYERS 2
#define EPAD 32    // edge feature dim padded 23 -> 32 (one WMMA K step)
#define ASTR 40    // LDS tile stride in bf16: 80B -> 16B-aligned b128 loads,
                   // bank stride 20 -> conflict-free across 16 lanes

// ---------------------------------------------------------------------------
// Optional CDNA5 async direct-to-LDS staging (ASYNCcnt-tracked)
// ---------------------------------------------------------------------------
#if __has_builtin(__builtin_amdgcn_global_load_async_to_lds_b128) && \
    __has_builtin(__builtin_amdgcn_s_wait_asynccnt)
#define USE_ASYNC 1
#else
#define USE_ASYNC 0
#endif

#if USE_ASYNC
typedef __attribute__((address_space(1))) v4i gv4i;   // global-AS int4
typedef __attribute__((address_space(3))) v4i lv4i;   // LDS-AS int4
__device__ inline void async_b128(const bf16_t* g, bf16_t* l) {
  __builtin_amdgcn_global_load_async_to_lds_b128(
      (gv4i*)(const void*)g, (lv4i*)(void*)l, 0, 0);
}
__device__ inline void async_wait0() { __builtin_amdgcn_s_wait_asynccnt(0); }
#else
__device__ inline void async_wait0() {}
#endif

__device__ inline void copy16_to_lds(const bf16_t* g, bf16_t* l) {
#if USE_ASYNC
  async_b128(g, l);
  async_b128(g + 8, l + 8);
#else
  const uint4* p = (const uint4*)g;
  uint4* d = (uint4*)l;
  d[0] = p[0]; d[1] = p[1];
#endif
}

// ---------------------------------------------------------------------------
// Fragment loaders: every fragment = two contiguous 16-byte (b128) loads.
// A (16x32 bf16, MxK), ISA 7.12.2: lane m=L&15, kb=(L&16)?8:0;
//   halves: K=kb..kb+7 and K=16+kb..16+kb+7  (both contiguous runs)
// B (32x16 bf16, KxN): lane n=L&15, kb=(L&16)?8:0;
//   halves: K=kb..kb+7 and K=16+kb..16+kb+7 of column n (tile stored column-major)
// ---------------------------------------------------------------------------
union frag_u { v16bf v; uint4 q[2]; };

__device__ inline v16bf load_frag_a40(const bf16_t* __restrict__ sA, int lane, int row0) {
  int m  = lane & 15;
  int kb = (lane & 16) ? 8 : 0;
  const bf16_t* r = sA + (row0 + m) * ASTR;
  frag_u u;
  u.q[0] = *(const uint4*)(r + kb);
  u.q[1] = *(const uint4*)(r + 16 + kb);
  return u.v;
}

__device__ inline v16bf load_frag_b40(const bf16_t* __restrict__ sBt, int lane, int col0) {
  int n  = col0 + (lane & 15);
  int kb = (lane & 16) ? 8 : 0;
  const bf16_t* c = sBt + n * ASTR + kb;
  frag_u u;
  u.q[0] = *(const uint4*)(c);
  u.q[1] = *(const uint4*)(c + 16);
  return u.v;
}

// ---------------------------------------------------------------------------
// One-time weight transposes (f32 -> bf16), so B tiles stage as contiguous
// column loads. Wt: (L, N=768, K=768) with Wt[l][n][k] = W[l][k][n].
// ---------------------------------------------------------------------------
__global__ void transpose_w_bf16(const float* __restrict__ W, bf16_t* __restrict__ Wt,
                                 size_t total) {
  size_t i = (size_t)blockIdx.x * blockDim.x + threadIdx.x;
  if (i >= total) return;
  size_t l = i / ((size_t)H * H);
  size_t r = i % ((size_t)H * H);
  int n = (int)(r / H), k = (int)(r % H);
  Wt[i] = (bf16_t)W[l * H * H + (size_t)k * H + n];
}

// We: (L,23,768) -> WeT: (L,768,32) with zero K-padding
__global__ void transpose_we_bf16(const float* __restrict__ We, bf16_t* __restrict__ WeT,
                                  size_t total) {
  size_t i = (size_t)blockIdx.x * blockDim.x + threadIdx.x;
  if (i >= total) return;
  size_t l = i / ((size_t)H * EPAD);
  size_t r = i % ((size_t)H * EPAD);
  int n = (int)(r / EPAD), k = (int)(r % EPAD);
  WeT[i] = (k < 23) ? (bf16_t)We[l * 23 * H + (size_t)k * H + n] : (bf16_t)0.0f;
}

// ---------------------------------------------------------------------------
// edge_attr bf16 (E x 32): [cont(7) | rel_table[rel](16) | zeros(9)]
// ---------------------------------------------------------------------------
__global__ void build_edge_attr(const float* __restrict__ cont, const int* __restrict__ rel_ids,
                                const float* __restrict__ rel_table,
                                bf16_t* __restrict__ eattr, int E) {
  int idx = blockIdx.x * blockDim.x + threadIdx.x;
  if (idx >= E * EPAD) return;
  int e = idx >> 5, c = idx & 31;
  float v;
  if (c < 7)       v = cont[(size_t)e * 7 + c];
  else if (c < 23) v = rel_table[(size_t)rel_ids[e] * 16 + (c - 7)];
  else             v = 0.0f;
  eattr[idx] = (bf16_t)v;
}

// ---------------------------------------------------------------------------
// h0 = word_feats + sinusoidal 2D posemb  ->  bf16
// ---------------------------------------------------------------------------
__global__ void posemb(const float* __restrict__ wf, const float* __restrict__ boxes,
                       bf16_t* __restrict__ hbf, size_t total) {
  size_t idx = (size_t)blockIdx.x * blockDim.x + threadIdx.x;
  if (idx >= total) return;
  int n = (int)(idx / H), c = (int)(idx % H);
  int coord = c / 192, t = c % 192;
  int j = (t < 96) ? t : (t - 96);
  float invf = __expf(-((float)(2 * j) / 192.0f) * 9.210340371976184f); // 10000^(-2j/192)
  float arg  = boxes[(size_t)n * 4 + coord] * invf;
  float pe   = (t < 96) ? __sinf(arg) : __cosf(arg);
  hbf[idx] = (bf16_t)(wf[idx] + pe);
}

// ---------------------------------------------------------------------------
// Node GEMM: out = h_bf16 @ W + bias. 128x128 tile/block, 8 waves, each wave
// 16 rows x 8 WMMA col-tiles. Double-buffered LDS pipeline: stage K-tile i+1
// (async direct-to-LDS where available) while WMMAs consume tile i; one
// barrier per K-step. A-rows clamped instead of guarded (OOB rows are never
// stored), so staging has no exec-mask branches.
// ---------------------------------------------------------------------------
__global__ __launch_bounds__(256)
void gemm_node(const bf16_t* __restrict__ A,
               const bf16_t* __restrict__ Wt0, const bf16_t* __restrict__ Wt1,
               const float* __restrict__ b0, const float* __restrict__ b1,
               float* __restrict__ o0, float* __restrict__ o1, int nrows) {
  __shared__ bf16_t sA[2][128 * ASTR];   // 2 x 10 KB
  __shared__ bf16_t sB[2][128 * ASTR];   // 2 x 10 KB (column-major: sB[n][k])

  const bf16_t* Wt   = blockIdx.z ? Wt1 : Wt0;
  const float*  bias = blockIdx.z ? b1 : b0;
  float*        out  = blockIdx.z ? o1 : o0;

  int row0 = blockIdx.y * 128;
  int col0 = blockIdx.x * 128;
  int tid  = threadIdx.x;
  int wave = tid >> 5, lane = tid & 31;

  const int half16 = (tid & 1) * 16;   // 0 or 16 elements
  const int pair   = tid >> 1;         // 0..127
  const int garow  = min(row0 + pair, nrows - 1);   // clamped A row
  const bf16_t* gA = A  + (size_t)garow * H + half16;
  const bf16_t* gB = Wt + (size_t)(col0 + pair) * H + half16;

  v8f acc[8];
#pragma unroll
  for (int t = 0; t < 8; ++t) acc[t] = (v8f){0.f,0.f,0.f,0.f,0.f,0.f,0.f,0.f};

  // prologue: stage K-tile 0 into buffer 0
  copy16_to_lds(gA, sA[0] + pair * ASTR + half16);
  copy16_to_lds(gB, sB[0] + pair * ASTR + half16);
  async_wait0();
  __syncthreads();

  int buf = 0;
  for (int k0 = 0; k0 < H; k0 += 32, buf ^= 1) {
    int kn = k0 + 32;
    if (kn < H) {   // stage next tile into the other buffer
      copy16_to_lds(gA + kn, sA[buf ^ 1] + pair * ASTR + half16);
      copy16_to_lds(gB + kn, sB[buf ^ 1] + pair * ASTR + half16);
    }
    // preload all fragments, then issue 8 WMMAs back-to-back
    v16bf afrag = load_frag_a40(sA[buf], lane, wave * 16);
    v16bf bfr[8];
#pragma unroll
    for (int t = 0; t < 8; ++t) bfr[t] = load_frag_b40(sB[buf], lane, t * 16);
#pragma unroll
    for (int t = 0; t < 8; ++t)
      acc[t] = __builtin_amdgcn_wmma_f32_16x16x32_bf16(
          false, afrag, false, bfr[t], (short)0, acc[t], false, false);
    if (kn < H) async_wait0();
    __syncthreads();
  }

  // C/D layout: lane n = lane&15 ; VGPR v -> M = v + (lane>=16 ? 8 : 0)
  int hi = (lane & 16) ? 8 : 0;
  int nn = lane & 15;
  int rbase = row0 + wave * 16 + hi;
  if (row0 + 128 <= nrows) {           // fast path: whole tile in range
#pragma unroll
    for (int t = 0; t < 8; ++t) {
      int c = col0 + t * 16 + nn;
      float bv = bias[c];
#pragma unroll
      for (int v = 0; v < 8; ++v)
        out[(size_t)(rbase + v) * H + c] = acc[t][v] + bv;
    }
  } else {
#pragma unroll
    for (int t = 0; t < 8; ++t) {
      int c = col0 + t * 16 + nn;
      float bv = bias[c];
#pragma unroll
      for (int v = 0; v < 8; ++v)
        if (rbase + v < nrows) out[(size_t)(rbase + v) * H + c] = acc[t][v] + bv;
    }
  }
}

// ---------------------------------------------------------------------------
// Fused edge kernel: e = edge_attr @ We (WMMA), gather xl[src]/xr[dst],
// leaky-relu, dot with att -> alpha. One wave per 16-edge tile.
// WeT (768 cols x 32 K) staged column-major in padded LDS (60 KB / 320 KB WGP).
// ---------------------------------------------------------------------------
__global__ __launch_bounds__(256)
void edge_alpha(const bf16_t* __restrict__ eattr, const bf16_t* __restrict__ WeT,
                const float* __restrict__ xl, const float* __restrict__ xr,
                const int* __restrict__ src, const int* __restrict__ dst,
                const float* __restrict__ att, float* __restrict__ alpha, int E) {
  __shared__ bf16_t sWe[H * ASTR];   // 61440 bytes
  for (int col = threadIdx.x; col < H; col += 256) {
    const bf16_t* s = WeT + (size_t)col * EPAD;
    bf16_t* d = sWe + col * ASTR;
    copy16_to_lds(s, d);
    copy16_to_lds(s + 16, d + 16);
  }
  async_wait0();
  __syncthreads();

  int wave = threadIdx.x >> 5, lane = threadIdx.x & 31;
  int e0 = (blockIdx.x * 8 + wave) * 16;
  if (e0 >= E) return;

  int hi = (lane & 16) ? 8 : 0;
  int nn = lane & 15;

  // A fragment straight from global (16 edges x 32 feats): 2 x b128
  v16bf afrag;
  {
    int er = min(e0 + nn, E - 1);
    const bf16_t* r = eattr + (size_t)er * EPAD;
    frag_u u;
    u.q[0] = *(const uint4*)(r + hi);          // K = kb..kb+7   (kb = hi = 0/8)
    u.q[1] = *(const uint4*)(r + 16 + hi);     // K = 16+kb..+7
    afrag = u.v;
  }

  int se[8], de[8];
#pragma unroll
  for (int v = 0; v < 8; ++v) {
    int e = min(e0 + hi + v, E - 1);
    se[v] = src[e]; de[v] = dst[e];
  }

  float accv[8] = {0.f,0.f,0.f,0.f,0.f,0.f,0.f,0.f};
  v16bf bcur = load_frag_b40(sWe, lane, 0);
  for (int ct = 0; ct < H / 16; ++ct) {
    v16bf bnext = (ct + 1 < H / 16) ? load_frag_b40(sWe, lane, (ct + 1) * 16) : bcur;
    v8f c = (v8f){0.f,0.f,0.f,0.f,0.f,0.f,0.f,0.f};
    c = __builtin_amdgcn_wmma_f32_16x16x32_bf16(
        false, afrag, false, bcur, (short)0, c, false, false);
    int ch = ct * 16 + nn;
    float aw = att[ch];
#pragma unroll
    for (int v = 0; v < 8; ++v) {
      float val = c[v] + xl[(size_t)se[v] * H + ch] + xr[(size_t)de[v] * H + ch];
      val = (val > 0.0f) ? val : 0.2f * val;    // leaky_relu(0.2)
      accv[v] += val * aw;
    }
    bcur = bnext;
  }
  // reduce over the 16 lanes of each half-wave (channels) -> alpha per edge
#pragma unroll
  for (int v = 0; v < 8; ++v) {
    float s = accv[v];
    s += __shfl_xor(s, 1, 32);
    s += __shfl_xor(s, 2, 32);
    s += __shfl_xor(s, 4, 32);
    s += __shfl_xor(s, 8, 32);
    if (nn == 0) {
      int e = e0 + hi + v;
      if (e < E) alpha[e] = s;
    }
  }
}

// ---------------------------------------------------------------------------
// Segment softmax over dst: order-preserving uint keys for atomic max.
// ---------------------------------------------------------------------------
__device__ inline unsigned fkey(float f) {
  unsigned u = __float_as_uint(f);
  return (u & 0x80000000u) ? ~u : (u | 0x80000000u);
}
__device__ inline float funkey(unsigned k) {
  unsigned u = (k & 0x80000000u) ? (k & 0x7fffffffu) : ~k;
  return __uint_as_float(u);
}

__global__ void zero_f32(float* p, size_t n) {
  size_t i = (size_t)blockIdx.x * blockDim.x + threadIdx.x;
  if (i < n) p[i] = 0.0f;
}
__global__ void zero_u32(unsigned* p, size_t n) {
  size_t i = (size_t)blockIdx.x * blockDim.x + threadIdx.x;
  if (i < n) p[i] = 0u;
}

__global__ void edge_max(const float* __restrict__ alpha, const int* __restrict__ dst,
                         unsigned* __restrict__ amax, int E) {
  int e = blockIdx.x * blockDim.x + threadIdx.x;
  if (e >= E) return;
  atomicMax(&amax[dst[e]], fkey(alpha[e]));
}

__global__ void edge_exp(float* __restrict__ alpha, const int* __restrict__ dst,
                         const unsigned* __restrict__ amax, float* __restrict__ denom, int E) {
  int e = blockIdx.x * blockDim.x + threadIdx.x;
  if (e >= E) return;
  float mx = funkey(amax[dst[e]]);
  float ex = __expf(alpha[e] - mx);
  alpha[e] = ex;   // reuse buffer as ex
  __hip_atomic_fetch_add(&denom[dst[e]], ex, __ATOMIC_RELAXED, __HIP_MEMORY_SCOPE_AGENT);
}

// one wave per edge: agg[dst] += a * xl[src]
__global__ __launch_bounds__(256)
void edge_scatter(const float* __restrict__ ex, const float* __restrict__ denom,
                  const int* __restrict__ src, const int* __restrict__ dst,
                  const float* __restrict__ xl, float* __restrict__ agg, int E) {
  int wave = threadIdx.x >> 5, lane = threadIdx.x & 31;
  int e = blockIdx.x * 8 + wave;
  if (e >= E) return;
  int s = src[e], d = dst[e];
  float a = ex[e] / (denom[d] + 1e-16f);
  const float* xs = xl + (size_t)s * H;
  float*       o  = agg + (size_t)d * H;
  for (int c = lane; c < H; c += 32)
    __hip_atomic_fetch_add(&o[c], a * xs[c], __ATOMIC_RELAXED, __HIP_MEMORY_SCOPE_AGENT);
}

// h = relu(agg + bias) -> bf16 (and optional fp32 final output)
__global__ void relu_bias(const float* __restrict__ agg, const float* __restrict__ bias,
                          bf16_t* __restrict__ hbf, float* __restrict__ outf, size_t total) {
  size_t idx = (size_t)blockIdx.x * blockDim.x + threadIdx.x;
  if (idx >= total) return;
  int c = (int)(idx % H);
  float v = agg[idx] + bias[c];
  v = (v > 0.0f) ? v : 0.0f;
  hbf[idx] = (bf16_t)v;
  if (outf) outf[idx] = v;
}

// ---------------------------------------------------------------------------
// Host-side orchestration
// ---------------------------------------------------------------------------
static inline size_t align256(size_t x) { return (x + 255) & ~(size_t)255; }

extern "C" void kernel_launch(void* const* d_in, const int* in_sizes, int n_in,
                              void* d_out, int out_size, void* d_ws, size_t ws_size,
                              hipStream_t stream) {
  const float* word_feats = (const float*)d_in[0];
  const float* boxes_norm = (const float*)d_in[1];
  const int*   edge_index = (const int*)  d_in[2];
  const float* edge_cont  = (const float*)d_in[3];
  const int*   rel_ids    = (const int*)  d_in[4];
  const float* rel_table  = (const float*)d_in[5];
  const float* Wl         = (const float*)d_in[6];
  const float* bl         = (const float*)d_in[7];
  const float* Wr         = (const float*)d_in[8];
  const float* br         = (const float*)d_in[9];
  const float* We         = (const float*)d_in[10];
  const float* att        = (const float*)d_in[11];
  const float* bias_out   = (const float*)d_in[12];

  const int N = in_sizes[0] / H;
  const int E = in_sizes[4];
  const int* srcI = edge_index;
  const int* dstI = edge_index + E;

  // workspace layout
  char* ws = (char*)d_ws;
  size_t off = 0;
  auto carve = [&](size_t bytes) { void* p = ws + off; off = align256(off + bytes); return p; };
  bf16_t*   h_bf  = (bf16_t*) carve((size_t)N * H * 2);
  float*    agg   = (float*)  carve((size_t)N * H * 4);
  float*    xl    = (float*)  carve((size_t)N * H * 4);
  float*    xr    = (float*)  carve((size_t)N * H * 4);
  bf16_t*   WlT   = (bf16_t*) carve((size_t)LAYERS * H * H * 2);
  bf16_t*   WrT   = (bf16_t*) carve((size_t)LAYERS * H * H * 2);
  bf16_t*   WeT   = (bf16_t*) carve((size_t)LAYERS * H * EPAD * 2);
  bf16_t*   eattr = (bf16_t*) carve((size_t)E * EPAD * 2);
  float*    alpha = (float*)  carve((size_t)E * 4);
  float*    denom = (float*)  carve((size_t)N * 4);
  unsigned* amax  = (unsigned*)carve((size_t)N * 4);
  (void)ws_size; (void)n_in; (void)out_size;

  const size_t totNH = (size_t)N * H;

  // 1. one-time weight transposes (f32 -> bf16)
  {
    size_t nW = (size_t)LAYERS * H * H;
    transpose_w_bf16<<<(int)((nW + 255) / 256), 256, 0, stream>>>(Wl, WlT, nW);
    transpose_w_bf16<<<(int)((nW + 255) / 256), 256, 0, stream>>>(Wr, WrT, nW);
    size_t nWe = (size_t)LAYERS * H * EPAD;
    transpose_we_bf16<<<(int)((nWe + 255) / 256), 256, 0, stream>>>(We, WeT, nWe);
  }
  // 2. edge attributes
  build_edge_attr<<<(E * EPAD + 255) / 256, 256, 0, stream>>>(edge_cont, rel_ids, rel_table, eattr, E);
  // 3. h0 = word + posemb (bf16)
  posemb<<<(int)((totNH + 255) / 256), 256, 0, stream>>>(word_feats, boxes_norm, h_bf, totNH);

  for (int l = 0; l < LAYERS; ++l) {
    const bf16_t* WlT_l = WlT + (size_t)l * H * H;
    const bf16_t* WrT_l = WrT + (size_t)l * H * H;
    const bf16_t* WeT_l = WeT + (size_t)l * H * EPAD;
    const float*  bl_l  = bl + l * H;
    const float*  br_l  = br + l * H;
    const float*  at_l  = att + l * H;
    const float*  bo_l  = bias_out + l * H;

    // node projections xl, xr via WMMA GEMM
    dim3 ggrid(H / 128, (N + 127) / 128, 2);
    gemm_node<<<ggrid, 256, 0, stream>>>(h_bf, WlT_l, WrT_l, bl_l, br_l, xl, xr, N);

    // reset accumulators
    zero_f32<<<(int)((totNH + 255) / 256), 256, 0, stream>>>(agg, totNH);
    zero_f32<<<(N + 255) / 256, 256, 0, stream>>>(denom, (size_t)N);
    zero_u32<<<(N + 255) / 256, 256, 0, stream>>>(amax, (size_t)N);

    // fused edge GEMM + gather + attention score
    edge_alpha<<<(E + 127) / 128, 256, 0, stream>>>(eattr, WeT_l, xl, xr, srcI, dstI, at_l, alpha, E);

    // segment softmax over dst
    edge_max<<<(E + 255) / 256, 256, 0, stream>>>(alpha, dstI, amax, E);
    edge_exp<<<(E + 255) / 256, 256, 0, stream>>>(alpha, dstI, amax, denom, E);

    // weighted aggregation
    edge_scatter<<<(E + 7) / 8, 256, 0, stream>>>(alpha, denom, srcI, dstI, xl, agg, E);

    // relu + bias; last layer also writes fp32 output
    float* outf = (l == LAYERS - 1) ? (float*)d_out : nullptr;
    relu_bias<<<(int)((totNH + 255) / 256), 256, 0, stream>>>(agg, bo_l, h_bf, outf, totNH);
  }
}
